// SpanBERT_CRF_87196426043987
// MI455X (gfx1250) — matrix operand order repaired and checked by reading, not compiled, not run on hardware
//
#include <hip/hip_runtime.h>
#include <math.h>

// ---------------------------------------------------------------------------
// Problem constants (match reference)
// ---------------------------------------------------------------------------
constexpr int kB = 4, kS = 512, kH = 768, kNH = 12, kDH = 64, kFF = 3072, kL = 6;
constexpr int kBS = kB * kS;              // 2048 tokens

// ---------------------------------------------------------------------------
// WMMA fragment types (gfx1250, wave32)
// ---------------------------------------------------------------------------
typedef __attribute__((ext_vector_type(16))) __bf16        v16bf;
typedef __attribute__((ext_vector_type(8)))  float         v8f;
typedef __attribute__((ext_vector_type(4)))  unsigned int  u32x4;
typedef __attribute__((ext_vector_type(4)))  int           v4i;
typedef __attribute__((ext_vector_type(4)))  float         f32x4;
typedef __attribute__((ext_vector_type(4)))  unsigned short u16x4;

union Frag { u32x4 u[2]; v16bf v; };

static __device__ inline unsigned short f32_to_bf16(float f) {
  unsigned int u = __builtin_bit_cast(unsigned int, f);
  u += 0x7FFFu + ((u >> 16) & 1u);       // round-to-nearest-even
  return (unsigned short)(u >> 16);
}

static __device__ inline v8f wmma_bf16(const v16bf& a, const v16bf& b, const v8f& c) {
  return __builtin_amdgcn_wmma_f32_16x16x32_bf16(false, a, false, b, (short)0, c,
                                                 false, false);
}

static __device__ inline float block_sum_256(float v, float* sh) {
  int t = threadIdx.x;
  sh[t] = v; __syncthreads();
  #pragma unroll
  for (int s = 128; s > 0; s >>= 1) {
    if (t < s) sh[t] += sh[t + s];
    __syncthreads();
  }
  float r = sh[0];
  __syncthreads();
  return r;
}

// ---------------------------------------------------------------------------
// Async global->LDS copy path (CDNA5; gated so absence falls back cleanly)
// ---------------------------------------------------------------------------
#if defined(__has_builtin)
#  if __has_builtin(__builtin_amdgcn_global_load_async_to_lds_b128) && \
      __has_builtin(__builtin_amdgcn_s_wait_asynccnt)
#    define HAVE_ASYNC_LDS 1
#  endif
#endif

#ifdef HAVE_ASYNC_LDS
static __device__ inline void async_copy_b128(const unsigned short* g,
                                              unsigned short* l) {
  __builtin_amdgcn_global_load_async_to_lds_b128(
      (__attribute__((address_space(1))) v4i*)(g),
      (__attribute__((address_space(3))) v4i*)(l),
      0, 0);
}
#endif

// ---------------------------------------------------------------------------
// f32 -> bf16 cast (activations, per layer; weights use transpose-cast below)
// ---------------------------------------------------------------------------
__global__ __launch_bounds__(256)
void cast_f32_bf16_kernel(const float* __restrict__ src,
                          unsigned short* __restrict__ dst, size_t n) {
  size_t i = (size_t)blockIdx.x * blockDim.x + threadIdx.x;
  size_t stride = (size_t)gridDim.x * blockDim.x;
  for (; i < n; i += stride) dst[i] = f32_to_bf16(src[i]);
}

// ---------------------------------------------------------------------------
// Weight cast + transpose: src f32 [L][K][N] -> dst bf16 [L][N][K].
// Tiled 32x32 through LDS, coalesced on both sides.  grid=(N/32, K/32, L).
// ---------------------------------------------------------------------------
__global__ __launch_bounds__(256)
void transpose_cast_kernel(const float* __restrict__ src,
                           unsigned short* __restrict__ dst, int K, int N) {
  __shared__ unsigned short tile[32][33];
  src += (size_t)blockIdx.z * K * N;
  dst += (size_t)blockIdx.z * N * K;
  const int n0 = blockIdx.x * 32, k0 = blockIdx.y * 32;
  const int tx = threadIdx.x & 31, ty = threadIdx.x >> 5;   // 32 x 8
  #pragma unroll
  for (int i = 0; i < 4; ++i) {
    int k = k0 + ty + i * 8;
    tile[ty + i * 8][tx] = f32_to_bf16(src[(size_t)k * N + n0 + tx]);
  }
  __syncthreads();
  #pragma unroll
  for (int i = 0; i < 4; ++i) {
    int n = n0 + ty + i * 8;
    dst[(size_t)n * K + k0 + tx] = tile[tx][ty + i * 8];
  }
}

// ---------------------------------------------------------------------------
// Embedding gather + LayerNorm.  grid = kBS tokens, block = 256, H = 768.
// ---------------------------------------------------------------------------
__global__ __launch_bounds__(256)
void embed_ln_kernel(const int* __restrict__ ids, const int* __restrict__ tts,
                     const float* __restrict__ we, const float* __restrict__ pe,
                     const float* __restrict__ te, const float* __restrict__ g,
                     const float* __restrict__ bb, float* __restrict__ x) {
  __shared__ float sh[256];
  const int token = blockIdx.x;
  const int s  = token % kS;
  const int id = ids[token];
  const int tt = tts[token];
  float v[3]; float sum = 0.f;
  #pragma unroll
  for (int i = 0; i < 3; ++i) {
    int idx = threadIdx.x + i * 256;
    v[i] = we[(size_t)id * kH + idx] + pe[(size_t)s * kH + idx] +
           te[(size_t)tt * kH + idx];
    sum += v[i];
  }
  float mu = block_sum_256(sum, sh) * (1.f / kH);
  float var = 0.f;
  #pragma unroll
  for (int i = 0; i < 3; ++i) { float d = v[i] - mu; var += d * d; }
  var = block_sum_256(var, sh) * (1.f / kH);
  float inv = rsqrtf(var + 1e-12f);
  #pragma unroll
  for (int i = 0; i < 3; ++i) {
    int idx = threadIdx.x + i * 256;
    x[(size_t)token * kH + idx] = (v[i] - mu) * inv * g[idx] + bb[idx];
  }
}

// ---------------------------------------------------------------------------
// Residual add + LayerNorm (in-place on x).  grid = kBS, block = 256.
// ---------------------------------------------------------------------------
__global__ __launch_bounds__(256)
void ln_residual_kernel(float* __restrict__ x, const float* __restrict__ t,
                        const float* __restrict__ g, const float* __restrict__ bb) {
  __shared__ float sh[256];
  const int token = blockIdx.x;
  float v[3]; float sum = 0.f;
  #pragma unroll
  for (int i = 0; i < 3; ++i) {
    int idx = threadIdx.x + i * 256;
    v[i] = x[(size_t)token * kH + idx] + t[(size_t)token * kH + idx];
    sum += v[i];
  }
  float mu = block_sum_256(sum, sh) * (1.f / kH);
  float var = 0.f;
  #pragma unroll
  for (int i = 0; i < 3; ++i) { float d = v[i] - mu; var += d * d; }
  var = block_sum_256(var, sh) * (1.f / kH);
  float inv = rsqrtf(var + 1e-12f);
  #pragma unroll
  for (int i = 0; i < 3; ++i) {
    int idx = threadIdx.x + i * 256;
    x[(size_t)token * kH + idx] = (v[i] - mu) * inv * g[idx] + bb[idx];
  }
}

// ---------------------------------------------------------------------------
// WMMA GEMM: C[M,N] = act(A[M,K] * B[K,N] + bias), bf16 in, f32 accumulate.
//   A  : bf16 activations, row-major M x K
//   Bt : bf16 weights, TRANSPOSED row-major N x K
// Both tiles are pure copies -> async global->LDS double-buffered pipeline
// (fallback: synchronous staging).  Block tile 128x128, K-step 32, 8 waves,
// each wave = 32x64 (2x4 WMMA tiles).  M%128==0, N%128==0, K%32==0.
// ---------------------------------------------------------------------------
constexpr int BM = 128, BN = 128, BK = 32;

__global__ __launch_bounds__(256)
void gemm_bf16_wmma(const unsigned short* __restrict__ A,
                    const unsigned short* __restrict__ Bt,
                    const float* __restrict__ bias, float* __restrict__ Cf,
                    unsigned short* __restrict__ Cb, int M, int N, int K,
                    int act_gelu) {
  __shared__ unsigned short As[2][BM][BK];   // [buf][row][k]
  __shared__ unsigned short Bs[2][BN][BK];   // [buf][col][k]

  const int tid      = threadIdx.x;
  const int wave     = tid >> 5;
  const int lane     = tid & 31;
  const int laneMod  = lane & 15;
  const int laneHalf = lane >> 4;
  const int waveM    = wave & 3;             // 0..3 -> 32-row slab
  const int waveN    = wave >> 2;            // 0..1 -> 64-col slab
  const int blockRow = blockIdx.y * BM;
  const int blockCol = blockIdx.x * BN;

  const v8f vzero = {0.f, 0.f, 0.f, 0.f, 0.f, 0.f, 0.f, 0.f};
  v8f acc[2][4];
  #pragma unroll
  for (int mt = 0; mt < 2; ++mt)
    #pragma unroll
    for (int nt = 0; nt < 4; ++nt) acc[mt][nt] = vzero;

  auto compute = [&](int buf) {
    Frag aF[2], bF[4];
    #pragma unroll
    for (int mt = 0; mt < 2; ++mt) {
      // A 16x32 layout: lane holds row M=lane%16; k = (i<8?i:i+8)+8*(lane/16)
      const u32x4* p = (const u32x4*)(&As[buf][waveM * 32 + mt * 16 + laneMod][0]);
      aF[mt].u[0] = p[laneHalf];
      aF[mt].u[1] = p[2 + laneHalf];
    }
    #pragma unroll
    for (int nt = 0; nt < 4; ++nt) {
      // B 32x16 layout: lane holds col N=lane%16; k = (i<8?i:i+8)+8*(lane/16)
      const u32x4* p = (const u32x4*)(&Bs[buf][waveN * 64 + nt * 16 + laneMod][0]);
      bF[nt].u[0] = p[laneHalf];
      bF[nt].u[1] = p[2 + laneHalf];
    }
    #pragma unroll
    for (int mt = 0; mt < 2; ++mt)
      #pragma unroll
      for (int nt = 0; nt < 4; ++nt)
        acc[mt][nt] = wmma_bf16(aF[mt].v, bF[nt].v, acc[mt][nt]);
  };

#ifdef HAVE_ASYNC_LDS
  // 4 async b128 ops per wave per buffer (2 for A, 2 for B); ASYNCcnt is
  // per-wave and completes in order, so waiting <=4 guarantees the previous
  // buffer's copies have landed.
  auto stage = [&](int buf, int k0) {
    #pragma unroll
    for (int i = 0; i < 2; ++i) {
      int c = tid + i * 256;                 // 512 x 16B chunks for A tile
      int row = c >> 2, seg = (c & 3) << 3;
      async_copy_b128(A + (size_t)(blockRow + row) * K + k0 + seg,
                      &As[buf][row][seg]);
    }
    #pragma unroll
    for (int i = 0; i < 2; ++i) {
      int c = tid + i * 256;                 // 512 x 16B chunks for B tile
      int col = c >> 2, seg = (c & 3) << 3;
      async_copy_b128(Bt + (size_t)(blockCol + col) * K + k0 + seg,
                      &Bs[buf][col][seg]);
    }
  };
  const int nIter = K / BK;
  stage(0, 0);
  for (int it = 0; it < nIter; ++it) {
    const int buf = it & 1;
    if (it + 1 < nIter) {
      stage(buf ^ 1, (it + 1) * BK);
      __builtin_amdgcn_s_wait_asynccnt(4);   // current buf's copies done
    } else {
      __builtin_amdgcn_s_wait_asynccnt(0);
    }
    __syncthreads();
    compute(buf);
    __syncthreads();                         // protect buf before restage
  }
#else
  for (int k0 = 0; k0 < K; k0 += BK) {
    if (k0 + BK < K) {
      __builtin_prefetch((const void*)(A + (size_t)(blockRow + (tid >> 2)) * K + k0 + BK), 0, 1);
      __builtin_prefetch((const void*)(Bt + (size_t)(blockCol + (tid >> 2)) * K + k0 + BK), 0, 1);
    }
    #pragma unroll
    for (int i = 0; i < 2; ++i) {
      int c = tid + i * 256;
      int row = c >> 2, seg = (c & 3) << 3;
      *(u32x4*)&As[0][row][seg] =
          *(const u32x4*)(A + (size_t)(blockRow + row) * K + k0 + seg);
    }
    #pragma unroll
    for (int i = 0; i < 2; ++i) {
      int c = tid + i * 256;
      int col = c >> 2, seg = (c & 3) << 3;
      *(u32x4*)&Bs[0][col][seg] =
          *(const u32x4*)(Bt + (size_t)(blockCol + col) * K + k0 + seg);
    }
    __syncthreads();
    compute(0);
    __syncthreads();
  }
#endif

  // --- epilogue: bias (+exact gelu), f32 or bf16 store
  // C/D layout: element j -> M = j + 8*(lane/16), N = lane%16
  #pragma unroll
  for (int nt = 0; nt < 4; ++nt) {
    int col = blockCol + waveN * 64 + nt * 16 + laneMod;
    float bv = bias[col];
    #pragma unroll
    for (int mt = 0; mt < 2; ++mt) {
      #pragma unroll
      for (int j = 0; j < 8; ++j) {
        int row = blockRow + waveM * 32 + mt * 16 + laneHalf * 8 + j;
        float v = acc[mt][nt][j] + bv;
        if (act_gelu) v = 0.5f * v * (1.f + erff(v * 0.70710678118f));
        if (Cb) Cb[(size_t)row * N + col] = f32_to_bf16(v);
        else    Cf[(size_t)row * N + col] = v;
      }
    }
  }
}

// ---------------------------------------------------------------------------
// Attention scores: S[b,h,q,k] = scale * q.k + mask_bias, via WMMA over DH=64.
// grid = (S/16/8, S/16, B*NH), block = 256 (8 waves, one 16x16 k-tile each).
// ---------------------------------------------------------------------------
__global__ __launch_bounds__(256)
void attn_scores_kernel(const unsigned short* __restrict__ qbf,
                        const unsigned short* __restrict__ kbf,
                        const int* __restrict__ amask,
                        float* __restrict__ scores, float scale) {
  const int tid = threadIdx.x;
  const int wave = tid >> 5, lane = tid & 31;
  const int laneMod = lane & 15, laneHalf = lane >> 4;
  const int bh = blockIdx.z;
  const int b = bh / kNH, h = bh % kNH;
  const int qtile = blockIdx.y;
  const int ktile = blockIdx.x * 8 + wave;

  const int qpos = qtile * 16 + laneMod;
  const int kpos = ktile * 16 + laneMod;
  const unsigned short* qrow = qbf + ((size_t)(b * kS + qpos) * kNH + h) * kDH;
  const unsigned short* krow = kbf + ((size_t)(b * kS + kpos) * kNH + h) * kDH;

  const v8f vzero = {0.f, 0.f, 0.f, 0.f, 0.f, 0.f, 0.f, 0.f};
  v8f acc = vzero;
  #pragma unroll
  for (int kc = 0; kc < 2; ++kc) {          // DH=64 -> two k=32 steps
    Frag aF, bF;
    aF.u[0] = *(const u32x4*)(qrow + kc * 32 + 8 * laneHalf);
    aF.u[1] = *(const u32x4*)(qrow + kc * 32 + 16 + 8 * laneHalf);
    bF.u[0] = *(const u32x4*)(krow + kc * 32 + 8 * laneHalf);
    bF.u[1] = *(const u32x4*)(krow + kc * 32 + 16 + 8 * laneHalf);
    acc = wmma_bf16(aF.v, bF.v, acc);
  }
  const float biasv = (1.f - (float)amask[b * kS + kpos]) * -1e9f;
  float* sbase = scores + (size_t)bh * kS * kS;
  #pragma unroll
  for (int j = 0; j < 8; ++j) {
    int qr = qtile * 16 + laneHalf * 8 + j;
    sbase[(size_t)qr * kS + kpos] = acc[j] * scale + biasv;
  }
}

// ---------------------------------------------------------------------------
// Row softmax (512 wide), writes bf16 probabilities.  grid = B*NH*S rows.
// ---------------------------------------------------------------------------
__global__ __launch_bounds__(256)
void softmax_kernel(const float* __restrict__ scores,
                    unsigned short* __restrict__ pbf) {
  __shared__ float sh[256];
  const int row = blockIdx.x;
  const int t = threadIdx.x;
  const float* src = scores + (size_t)row * kS;
  float a = src[t], c = src[t + 256];
  sh[t] = fmaxf(a, c); __syncthreads();
  #pragma unroll
  for (int s = 128; s > 0; s >>= 1) {
    if (t < s) sh[t] = fmaxf(sh[t], sh[t + s]);
    __syncthreads();
  }
  float m = sh[0]; __syncthreads();
  float e0 = expf(a - m), e1 = expf(c - m);
  sh[t] = e0 + e1; __syncthreads();
  #pragma unroll
  for (int s = 128; s > 0; s >>= 1) {
    if (t < s) sh[t] += sh[t + s];
    __syncthreads();
  }
  float inv = 1.f / sh[0];
  pbf[(size_t)row * kS + t]       = f32_to_bf16(e0 * inv);
  pbf[(size_t)row * kS + t + 256] = f32_to_bf16(e1 * inv);
}

// ---------------------------------------------------------------------------
// O = P @ V via WMMA.  grid = (S/16/8, B*NH), block = 256.
// Each wave owns one 16(q) x 64(dh) tile; V tiles staged transposed in LDS.
// Output written as bf16 in [B,S,NH,DH] (feeds the Wo GEMM directly).
// ---------------------------------------------------------------------------
__global__ __launch_bounds__(256)
void attn_out_kernel(const unsigned short* __restrict__ pbf,
                     const unsigned short* __restrict__ vbf,
                     unsigned short* __restrict__ obf) {
  __shared__ unsigned short vs[kDH][32];    // [dh][key-within-tile]
  const int tid = threadIdx.x;
  const int wave = tid >> 5, lane = tid & 31;
  const int laneMod = lane & 15, laneHalf = lane >> 4;
  const int bh = blockIdx.y;
  const int b = bh / kNH, h = bh % kNH;
  const int qtile = blockIdx.x * 8 + wave;
  const int qpos = qtile * 16 + laneMod;
  const unsigned short* prow = pbf + ((size_t)bh * kS + qpos) * kS;

  const v8f vzero = {0.f, 0.f, 0.f, 0.f, 0.f, 0.f, 0.f, 0.f};
  v8f acc[4];
  #pragma unroll
  for (int nt = 0; nt < 4; ++nt) acc[nt] = vzero;

  for (int kt = 0; kt < kS / 32; ++kt) {
    #pragma unroll
    for (int i = 0; i < 2; ++i) {
      int linear = tid + i * 256;
      int key = linear >> 4;
      int d4  = (linear & 15) << 2;
      u16x4 hv = *(const u16x4*)(vbf +
          ((size_t)(b * kS + kt * 32 + key) * kNH + h) * kDH + d4);
      vs[d4 + 0][key] = hv.x;
      vs[d4 + 1][key] = hv.y;
      vs[d4 + 2][key] = hv.z;
      vs[d4 + 3][key] = hv.w;
    }
    __syncthreads();
    Frag aF;
    aF.u[0] = *(const u32x4*)(prow + kt * 32 + 8 * laneHalf);
    aF.u[1] = *(const u32x4*)(prow + kt * 32 + 16 + 8 * laneHalf);
    #pragma unroll
    for (int nt = 0; nt < 4; ++nt) {
      Frag bF;
      const u32x4* p = (const u32x4*)(&vs[nt * 16 + laneMod][0]);
      bF.u[0] = p[laneHalf];
      bF.u[1] = p[2 + laneHalf];
      acc[nt] = wmma_bf16(aF.v, bF.v, acc[nt]);
    }
    __syncthreads();
  }
  #pragma unroll
  for (int nt = 0; nt < 4; ++nt) {
    int dh = nt * 16 + laneMod;
    #pragma unroll
    for (int j = 0; j < 8; ++j) {
      int qp = qtile * 16 + laneHalf * 8 + j;
      obf[((size_t)(b * kS + qp) * kNH + h) * kDH + dh] = f32_to_bf16(acc[nt][j]);
    }
  }
}

// ---------------------------------------------------------------------------
// QA head: emissions[token, 0/1] = x . qa_w[:,0/1] + qa_b.  grid = kBS.
// ---------------------------------------------------------------------------
__global__ __launch_bounds__(256)
void emissions_kernel(const float* __restrict__ x, const float* __restrict__ qa_w,
                      const float* __restrict__ qa_b, float* __restrict__ em) {
  __shared__ float s0[256], s1[256];
  const int token = blockIdx.x;
  float a0 = 0.f, a1 = 0.f;
  for (int i = threadIdx.x; i < kH; i += 256) {
    float xv = x[(size_t)token * kH + i];
    a0 += xv * qa_w[i * 2 + 0];
    a1 += xv * qa_w[i * 2 + 1];
  }
  s0[threadIdx.x] = a0; s1[threadIdx.x] = a1; __syncthreads();
  #pragma unroll
  for (int s = 128; s > 0; s >>= 1) {
    if (threadIdx.x < s) {
      s0[threadIdx.x] += s0[threadIdx.x + s];
      s1[threadIdx.x] += s1[threadIdx.x + s];
    }
    __syncthreads();
  }
  if (threadIdx.x == 0) {
    em[(size_t)token * 2 + 0] = s0[0] + qa_b[0];
    em[(size_t)token * 2 + 1] = s1[0] + qa_b[1];
  }
}

// ---------------------------------------------------------------------------
// 2-state CRF negative log-likelihood (sum over batch).  Trivial work.
// ---------------------------------------------------------------------------
static __device__ inline float logadd2(float x, float y) {
  float m = fmaxf(x, y);
  return m + logf(expf(x - m) + expf(y - m));
}

__global__ void crf_kernel(const float* __restrict__ em, const int* __restrict__ amask,
                           const int* __restrict__ sp, const int* __restrict__ ep,
                           const float* __restrict__ st, const float* __restrict__ en,
                           const float* __restrict__ tr, float* __restrict__ out) {
  __shared__ float llh[kB];
  const int t = threadIdx.x;
  if (t < kB) {
    int s0i = sp[t], e0i = ep[t];
    bool valid = (s0i >= 0) && (s0i < kS) && (e0i >= 0) && (e0i < kS) && (s0i <= e0i);
    const float* E = em + (size_t)t * kS * 2;
    const int* mk = amask + t * kS;
    int prev = (valid && 0 >= s0i && 0 <= e0i) ? 1 : 0;
    float num = st[prev] + E[prev];
    float a0 = st[0] + E[0];
    float a1 = st[1] + E[1];
    int cnt = mk[0] ? 1 : 0;
    float t00 = tr[0], t01 = tr[1], t10 = tr[2], t11 = tr[3];
    for (int p = 1; p < kS; ++p) {
      int cur = (valid && p >= s0i && p <= e0i) ? 1 : 0;
      int m = mk[p];
      if (m) {
        num += tr[prev * 2 + cur] + E[p * 2 + cur];
        cnt++;
      }
      float n0 = logadd2(a0 + t00, a1 + t10) + E[p * 2 + 0];
      float n1 = logadd2(a0 + t01, a1 + t11) + E[p * 2 + 1];
      if (m) { a0 = n0; a1 = n1; }
      prev = cur;
    }
    int lastIdx = cnt - 1;
    int lastTag = (valid && lastIdx >= s0i && lastIdx <= e0i) ? 1 : 0;
    num += en[lastTag];
    float logZ = logadd2(a0 + en[0], a1 + en[1]);
    llh[t] = num - logZ;
  }
  __syncthreads();
  if (t == 0) {
    float s = 0.f;
    for (int i = 0; i < kB; ++i) s += llh[i];
    out[0] = -s;
  }
}

// ---------------------------------------------------------------------------
// Host orchestration.  Workspace ~210 MB, carved deterministically.
// ---------------------------------------------------------------------------
extern "C" void kernel_launch(void* const* d_in, const int* in_sizes, int n_in,
                              void* d_out, int out_size, void* d_ws, size_t ws_size,
                              hipStream_t stream) {
  (void)in_sizes; (void)n_in; (void)out_size; (void)ws_size;

  const int*   input_ids  = (const int*)d_in[0];
  const int*   attn_mask  = (const int*)d_in[1];
  const int*   token_type = (const int*)d_in[2];
  const int*   start_pos  = (const int*)d_in[3];
  const int*   end_pos    = (const int*)d_in[4];
  const float* word_emb   = (const float*)d_in[5];
  const float* pos_emb    = (const float*)d_in[6];
  const float* type_emb   = (const float*)d_in[7];
  const float* emb_g      = (const float*)d_in[8];
  const float* emb_b      = (const float*)d_in[9];
  const float* Wq  = (const float*)d_in[10];
  const float* bq  = (const float*)d_in[11];
  const float* Wk  = (const float*)d_in[12];
  const float* bk  = (const float*)d_in[13];
  const float* Wv  = (const float*)d_in[14];
  const float* bv  = (const float*)d_in[15];
  const float* Wo  = (const float*)d_in[16];
  const float* bo  = (const float*)d_in[17];
  const float* ln1g = (const float*)d_in[18];
  const float* ln1b = (const float*)d_in[19];
  const float* W1  = (const float*)d_in[20];
  const float* b1  = (const float*)d_in[21];
  const float* W2  = (const float*)d_in[22];
  const float* b2  = (const float*)d_in[23];
  const float* ln2g = (const float*)d_in[24];
  const float* ln2b = (const float*)d_in[25];
  const float* qa_w = (const float*)d_in[26];
  const float* qa_b = (const float*)d_in[27];
  const float* crf_start = (const float*)d_in[28];
  const float* crf_end   = (const float*)d_in[29];
  const float* crf_trans = (const float*)d_in[30];

  char* wsp = (char*)d_ws;
  auto carve = [&](size_t bytes) {
    char* p = wsp;
    wsp += (bytes + 255) & ~(size_t)255;
    return p;
  };

  const size_t whh = (size_t)kH * kH;     // 589824
  const size_t whf = (size_t)kH * kFF;    // 2359296

  // transposed bf16 weights [N][K]
  unsigned short* wqT = (unsigned short*)carve((size_t)kL * whh * 2);
  unsigned short* wkT = (unsigned short*)carve((size_t)kL * whh * 2);
  unsigned short* wvT = (unsigned short*)carve((size_t)kL * whh * 2);
  unsigned short* woT = (unsigned short*)carve((size_t)kL * whh * 2);
  unsigned short* w1T = (unsigned short*)carve((size_t)kL * whf * 2);
  unsigned short* w2T = (unsigned short*)carve((size_t)kL * whf * 2);
  float*          x    = (float*)carve((size_t)kBS * kH * 4);
  unsigned short* x_bf = (unsigned short*)carve((size_t)kBS * kH * 2);
  unsigned short* q_bf = (unsigned short*)carve((size_t)kBS * kH * 2);
  unsigned short* k_bf = (unsigned short*)carve((size_t)kBS * kH * 2);
  unsigned short* v_bf = (unsigned short*)carve((size_t)kBS * kH * 2);
  unsigned short* o_bf = (unsigned short*)carve((size_t)kBS * kH * 2);
  unsigned short* h_bf = (unsigned short*)carve((size_t)kBS * kFF * 2);
  float*          scores = (float*)carve((size_t)kB * kNH * kS * kS * 4);
  unsigned short* p_bf   = (unsigned short*)carve((size_t)kB * kNH * kS * kS * 2);
  float*          tmp1 = (float*)carve((size_t)kBS * kH * 4);
  float*          tmp2 = (float*)carve((size_t)kBS * kH * 4);
  float*          em   = (float*)carve((size_t)kBS * 2 * 4);

  // ---- weight transpose-casts (once per launch; ~85 MB bf16 total)
  transpose_cast_kernel<<<dim3(kH / 32, kH / 32, kL), 256, 0, stream>>>(Wq, wqT, kH, kH);
  transpose_cast_kernel<<<dim3(kH / 32, kH / 32, kL), 256, 0, stream>>>(Wk, wkT, kH, kH);
  transpose_cast_kernel<<<dim3(kH / 32, kH / 32, kL), 256, 0, stream>>>(Wv, wvT, kH, kH);
  transpose_cast_kernel<<<dim3(kH / 32, kH / 32, kL), 256, 0, stream>>>(Wo, woT, kH, kH);
  transpose_cast_kernel<<<dim3(kFF / 32, kH / 32, kL), 256, 0, stream>>>(W1, w1T, kH, kFF);
  transpose_cast_kernel<<<dim3(kH / 32, kFF / 32, kL), 256, 0, stream>>>(W2, w2T, kFF, kH);

  // ---- embeddings + LN
  embed_ln_kernel<<<kBS, 256, 0, stream>>>(input_ids, token_type, word_emb,
                                           pos_emb, type_emb, emb_g, emb_b, x);

  const dim3 gH(kH / BN, kBS / BM);    // (6, 16)
  const dim3 gFF(kFF / BN, kBS / BM);  // (24, 16)
  const dim3 gScores((kS / 16) / 8, kS / 16, kB * kNH);  // (4, 32, 48)
  const dim3 gAttnO((kS / 16) / 8, kB * kNH);            // (4, 48)
  const float scale = 0.125f;          // 1/sqrt(DH=64)
  const size_t nXH = (size_t)kBS * kH;

  for (int l = 0; l < kL; ++l) {
    // QKV projections from bf16 x -> bf16 outputs, layout [B,S,NH,DH]
    cast_f32_bf16_kernel<<<1536, 256, 0, stream>>>(x, x_bf, nXH);
    gemm_bf16_wmma<<<gH, 256, 0, stream>>>(x_bf, wqT + l * whh, bq + l * kH,
                                           nullptr, q_bf, kBS, kH, kH, 0);
    gemm_bf16_wmma<<<gH, 256, 0, stream>>>(x_bf, wkT + l * whh, bk + l * kH,
                                           nullptr, k_bf, kBS, kH, kH, 0);
    gemm_bf16_wmma<<<gH, 256, 0, stream>>>(x_bf, wvT + l * whh, bv + l * kH,
                                           nullptr, v_bf, kBS, kH, kH, 0);
    // attention
    attn_scores_kernel<<<gScores, 256, 0, stream>>>(q_bf, k_bf, attn_mask,
                                                    scores, scale);
    softmax_kernel<<<kB * kNH * kS, 256, 0, stream>>>(scores, p_bf);
    attn_out_kernel<<<gAttnO, 256, 0, stream>>>(p_bf, v_bf, o_bf);
    // output projection + residual LN
    gemm_bf16_wmma<<<gH, 256, 0, stream>>>(o_bf, woT + l * whh, bo + l * kH,
                                           tmp1, nullptr, kBS, kH, kH, 0);
    ln_residual_kernel<<<kBS, 256, 0, stream>>>(x, tmp1, ln1g + l * kH,
                                                ln1b + l * kH);
    // FFN: gelu(x@W1+b1) (bf16 out) @ W2 + b2, residual LN
    cast_f32_bf16_kernel<<<1536, 256, 0, stream>>>(x, x_bf, nXH);
    gemm_bf16_wmma<<<gFF, 256, 0, stream>>>(x_bf, w1T + l * whf, b1 + l * kFF,
                                            nullptr, h_bf, kBS, kFF, kH, 1);
    gemm_bf16_wmma<<<gH, 256, 0, stream>>>(h_bf, w2T + l * whf, b2 + l * kH,
                                           tmp2, nullptr, kBS, kH, kFF, 0);
    ln_residual_kernel<<<kBS, 256, 0, stream>>>(x, tmp2, ln2g + l * kH,
                                                ln2b + l * kH);
  }

  emissions_kernel<<<kBS, 256, 0, stream>>>(x, qa_w, qa_b, em);
  crf_kernel<<<1, 32, 0, stream>>>(em, attn_mask, start_pos, end_pos,
                                   crf_start, crf_end, crf_trans, (float*)d_out);
}
